// Quantize_3204045602891
// MI455X (gfx1250) — compile-verified, matrix operand order checked by reading, not run on hardware
//
#include <hip/hip_runtime.h>

// ---- types for CDNA5 WMMA -------------------------------------------------
typedef __attribute__((ext_vector_type(16))) __bf16 v16bf;
typedef __attribute__((ext_vector_type(8)))  __bf16 v8bf;
typedef __attribute__((ext_vector_type(8)))  float  v8f;

// ---- problem constants (from reference: enc (32,64,64,64), embed (512,64))
#define DQ        64
#define KCODES    512
#define NROWS     131072          // 32*64*64*64 / 64
#define ROWS_PER_WG 128           // 8 waves * 16 rows
#define NWG       (NROWS / ROWS_PER_WG)   // 1024
#define PITCH     72              // LDS row pitch in bf16 elems (16B aligned)
#define QELEMS    8388608         // NROWS * DQ

__device__ __forceinline__ v16bf join8(v8bf a, v8bf b) {
  return __builtin_shufflevector(a, b, 0,1,2,3,4,5,6,7,8,9,10,11,12,13,14,15);
}

// B fragments (hi/lo split) for one 16-code tile: 2 K-slabs x {hi,lo}
struct BFrag { v16bf h0, l0, h1, l1; };

__device__ __forceinline__ BFrag load_bfrag(const __bf16* __restrict__ ehi,
                                            const __bf16* __restrict__ elo,
                                            int code, int koff) {
  const __bf16* bh = ehi + code * PITCH;
  const __bf16* bl = elo + code * PITCH;
  BFrag f;
  f.h0 = join8(*(const v8bf*)(bh + koff),      *(const v8bf*)(bh + 16 + koff));
  f.l0 = join8(*(const v8bf*)(bl + koff),      *(const v8bf*)(bl + 16 + koff));
  f.h1 = join8(*(const v8bf*)(bh + 32 + koff), *(const v8bf*)(bh + 48 + koff));
  f.l1 = join8(*(const v8bf*)(bl + 32 + koff), *(const v8bf*)(bl + 48 + koff));
  return f;
}

__global__ __launch_bounds__(256)
void vq_main(const float* __restrict__ enc, const float* __restrict__ embed,
             float* __restrict__ quant_out, float* __restrict__ closest_out,
             float* __restrict__ wg_partial) {
  extern __shared__ char smem[];
  __bf16* ehi = (__bf16*)smem;                     // KCODES*PITCH bf16
  __bf16* elo = ehi + KCODES * PITCH;              // KCODES*PITCH bf16
  float*  esq = (float*)(elo + KCODES * PITCH);    // KCODES floats
  float*  wls = esq + KCODES;                      // 8 floats

  const int tid = threadIdx.x;

  // ---- stage codebook into LDS as bf16 hi/lo split + e_sq ----
  for (int k = tid; k < KCODES; k += 256) {
    const float* src = embed + k * DQ;
    float ss = 0.f;
    #pragma unroll
    for (int d = 0; d < DQ; ++d) {
      float v = src[d];
      ss = fmaf(v, v, ss);
      __bf16 h = (__bf16)v;
      float rem = v - (float)h;
      ehi[k * PITCH + d] = h;
      elo[k * PITCH + d] = (__bf16)rem;
    }
    esq[k] = ss;
  }
  __syncthreads();

  const int wave = tid >> 5;
  const int lane = tid & 31;
  const int hl   = lane >> 4;          // half-wave select
  const int mloc = lane & 15;          // row (A) / col (B) within tile
  const int koff = hl * 8;             // K sub-offset per documented layout

  const int row_base = (blockIdx.x * 8 + wave) * 16;
  const float* xr = enc + (size_t)(row_base + mloc) * DQ;

  // ---- load x tile (each lane: 32 of its row's 64 values, per A layout) ----
  float c[4][8];
  #pragma unroll
  for (int j = 0; j < 4; ++j) {
    #pragma unroll
    for (int t = 0; t < 8; ++t) c[j][t] = xr[j * 16 + koff + t];
  }
  float xp = 0.f;
  #pragma unroll
  for (int j = 0; j < 4; ++j) {
    #pragma unroll
    for (int t = 0; t < 8; ++t) xp = fmaf(c[j][t], c[j][t], xp);
  }
  float xsq = xp + __shfl_xor(xp, 16);   // full row sum-of-squares

  // ---- build A fragments (hi/lo bf16 split) for the two K=32 slabs ----
  v16bf Ah[2], Al[2];
  #pragma unroll
  for (int s = 0; s < 2; ++s) {
    #pragma unroll
    for (int t = 0; t < 8; ++t) {
      float v0 = c[2 * s][t], v1 = c[2 * s + 1][t];
      __bf16 h0 = (__bf16)v0, h1 = (__bf16)v1;
      Ah[s][t]     = h0;                 Ah[s][8 + t] = h1;
      Al[s][t]     = (__bf16)(v0 - (float)h0);
      Al[s][8 + t] = (__bf16)(v1 - (float)h1);
    }
  }

  // ---- x_sq aligned to the C/D layout (reg r -> row r or r+8) ----
  float xsr[8];
  #pragma unroll
  for (int r = 0; r < 8; ++r) {
    float lo = __shfl(xsq, r);
    float hi = __shfl(xsq, r + 8);
    xsr[r] = hl ? hi : lo;
  }

  float minv[8]; int mini[8];
  #pragma unroll
  for (int r = 0; r < 8; ++r) { minv[r] = 3.4e38f; mini[r] = 0; }

  // ---- main loop: 32 codebook tiles, 6 WMMA each (bf16x3, 2 acc chains),
  //      software-pipelined B-fragment prefetch to hide LDS latency ----
  BFrag cur = load_bfrag(ehi, elo, mloc, koff);   // tile 0
  float en_cur = esq[mloc];

  for (int tile = 0; tile < 32; ++tile) {
    const int code      = tile * 16 + mloc;
    const int next_code = (((tile + 1) & 31) * 16) + mloc;

    // prefetch next tile's fragments while this tile's WMMAs run
    BFrag nxt      = load_bfrag(ehi, elo, next_code, koff);
    float en_next  = esq[next_code];

    v8f acc0, acc1;
    acc0 = __builtin_amdgcn_wmma_f32_16x16x32_bf16(false, Ah[0], false, cur.h0,
                                                   (short)0, v8f{}, false, false);
    acc1 = __builtin_amdgcn_wmma_f32_16x16x32_bf16(false, Al[0], false, cur.h0,
                                                   (short)0, v8f{}, false, false);
    acc0 = __builtin_amdgcn_wmma_f32_16x16x32_bf16(false, Ah[0], false, cur.l0,
                                                   (short)0, acc0, false, false);
    acc1 = __builtin_amdgcn_wmma_f32_16x16x32_bf16(false, Al[1], false, cur.h1,
                                                   (short)0, acc1, false, false);
    acc0 = __builtin_amdgcn_wmma_f32_16x16x32_bf16(false, Ah[1], false, cur.h1,
                                                   (short)0, acc0, false, false);
    acc0 = __builtin_amdgcn_wmma_f32_16x16x32_bf16(false, Ah[1], false, cur.l1,
                                                   (short)0, acc0, false, false);

    #pragma unroll
    for (int r = 0; r < 8; ++r) {
      float dot = acc0[r] + acc1[r];
      float t2  = fmaf(-2.f, dot, en_cur);   // e_sq - 2*dot (x_sq added later)
      if (t2 < minv[r]) { minv[r] = t2; mini[r] = code; }
    }

    cur = nxt;
    en_cur = en_next;
  }

  // ---- argmin reduce across the 16 lanes of each half-wave ----
  #pragma unroll
  for (int r = 0; r < 8; ++r) {
    #pragma unroll
    for (int s = 8; s >= 1; s >>= 1) {
      float ov = __shfl_xor(minv[r], s);
      int   oi = __shfl_xor(mini[r], s);
      if (ov < minv[r] || (ov == minv[r] && oi < mini[r])) {
        minv[r] = ov; mini[r] = oi;
      }
    }
  }

  // ---- epilogue: indices, loss, coalesced fp32 codebook gather ----
  float lsum = 0.f;
  #pragma unroll
  for (int r = 0; r < 8; ++r) {
    const int il = __shfl(mini[r], 0);    // winner for row row_base+r
    const int ih = __shfl(mini[r], 16);   // winner for row row_base+8+r
    if (lane == 0 || lane == 16) {
      lsum += xsr[r] + minv[r];           // = d2min for this lane's row
      closest_out[row_base + r + hl * 8] = (float)mini[r];
    }
    const float2* sl = (const float2*)(embed + (size_t)il * DQ);
    const float2* sh = (const float2*)(embed + (size_t)ih * DQ);
    float2* dl = (float2*)(quant_out + (size_t)(row_base + r) * DQ);
    float2* dh = (float2*)(quant_out + (size_t)(row_base + 8 + r) * DQ);
    dl[lane] = sl[lane];
    dh[lane] = sh[lane];
  }

  lsum += __shfl_xor(lsum, 16);
  if (lane == 0) wls[wave] = lsum;
  __syncthreads();
  if (tid == 0) {
    float s = 0.f;
    #pragma unroll
    for (int w = 0; w < 8; ++w) s += wls[w];
    wg_partial[blockIdx.x] = s;           // fixed-order, deterministic
  }
}

__global__ void vq_finalize(const float* __restrict__ wg_partial,
                            float* __restrict__ loss_out) {
  if (threadIdx.x == 0 && blockIdx.x == 0) {
    float s = 0.f;
    for (int i = 0; i < NWG; ++i) s += wg_partial[i];
    // quantize_loss = codebook + commitment = 2 * mean((q-x)^2)
    loss_out[0] = 2.0f * s / (float)QELEMS;
  }
}

extern "C" void kernel_launch(void* const* d_in, const int* in_sizes, int n_in,
                              void* d_out, int out_size, void* d_ws, size_t ws_size,
                              hipStream_t stream) {
  const float* enc   = (const float*)d_in[0];   // (32,64,64,64) fp32
  const float* embed = (const float*)d_in[1];   // (512,64) fp32
  float* out     = (float*)d_out;
  float* quant   = out;                 // 8388608 floats
  float* loss    = out + QELEMS;        // 1 float
  float* closest = out + QELEMS + 1;    // 131072 floats
  float* partial = (float*)d_ws;        // NWG floats of scratch

  const size_t shmem = (size_t)KCODES * PITCH * 2 * sizeof(__bf16)
                     + (size_t)KCODES * sizeof(float) + 8 * sizeof(float);

  vq_main<<<NWG, 256, shmem, stream>>>(enc, embed, quant, closest, partial);
  vq_finalize<<<1, 64, 0, stream>>>(partial, loss);
}
